// CompletelyFixedSAM2Attention_1657857376485
// MI455X (gfx1250) — matrix operand order repaired and checked by reading, not compiled, not run on hardware
//
#include <hip/hip_runtime.h>
#include <hip/hip_bf16.h>

typedef __attribute__((ext_vector_type(16))) _Float16 v16h;
typedef __attribute__((ext_vector_type(8)))  float    v8f;

#define BB   2
#define NTOK 4096      // 16*16*16 pooled tokens per batch
#define CCH  64
#define CQK  8
#define L2E  1.44269504088896f

// ---------------------------------------------------------------------------
// Kernel 1: pool x (4^3 mean) then project -> qf[B,4096,8], kf[B,4096,8] (f32),
//           vT[B,64,4096] (f16, transposed so WMMA B-operand loads are contiguous)
// 1 block per pooled token, 64 threads. Loads are float4 (b128) with a 4-way
// LDS partial reduction: 512B per wave-instruction on the 134MB x stream.
// ---------------------------------------------------------------------------
__global__ __launch_bounds__(64)
void pool_project_kernel(const float* __restrict__ x,
                         const float* __restrict__ Wq, const float* __restrict__ bq,
                         const float* __restrict__ Wk, const float* __restrict__ bk,
                         const float* __restrict__ Wv, const float* __restrict__ bv,
                         float* __restrict__ qf, float* __restrict__ kf,
                         _Float16* __restrict__ vT)
{
    __shared__ __align__(16) float4 part[4][16];
    __shared__ __align__(16) float  xp[64];

    const int t  = blockIdx.x;           // 0..B*4096-1
    const int b  = t >> 12;
    const int r  = t & (NTOK - 1);
    const int ph = r >> 8, pw = (r >> 4) & 15, pd = r & 15;
    const int tid = threadIdx.x;
    const int c4  = tid & 15;            // channel quad 0..15
    const int grp = tid >> 4;            // voxel group 0..3 (16 voxels each)

    const float* xb = x + ((size_t)b << 18) * CCH;  // b*64*64*64*64
    float4 s4 = make_float4(0.f, 0.f, 0.f, 0.f);
#pragma unroll
    for (int i = 0; i < 16; ++i) {
        const int v  = grp * 16 + i;
        const int dh = v >> 4, dw = (v >> 2) & 3, dd = v & 3;
        const size_t vox = ((size_t)(ph * 4 + dh) * 64 + (pw * 4 + dw)) * 64 + (pd * 4 + dd);
        const float4 xv = *(const float4*)(xb + vox * CCH + c4 * 4);
        s4.x += xv.x; s4.y += xv.y; s4.z += xv.z; s4.w += xv.w;
    }
    part[grp][c4] = s4;
    __syncthreads();
    if (grp == 0) {
        const float4 a = part[0][c4], b2 = part[1][c4], c2 = part[2][c4], d2 = part[3][c4];
        float4 m;
        m.x = (a.x + b2.x + c2.x + d2.x) * (1.0f / 64.0f);
        m.y = (a.y + b2.y + c2.y + d2.y) * (1.0f / 64.0f);
        m.z = (a.z + b2.z + c2.z + d2.z) * (1.0f / 64.0f);
        m.w = (a.w + b2.w + c2.w + d2.w) * (1.0f / 64.0f);
        *(float4*)(xp + c4 * 4) = m;
    }
    __syncthreads();

    // V projection (all 64 threads, one output channel each): vT[b][c][token]
    const int c = tid;
    float av = bv[c];
#pragma unroll 8
    for (int j = 0; j < 64; ++j) av += xp[j] * Wv[j * CCH + c];
    vT[((size_t)b * CCH + c) * NTOK + r] = (_Float16)av;

    if (c < CQK) {
        float a = bq[c];
#pragma unroll 8
        for (int j = 0; j < 64; ++j) a += xp[j] * Wq[j * CQK + c];
        qf[((size_t)b * NTOK + r) * CQK + c] = a;
    } else if (c < 2 * CQK) {
        const int c2 = c - CQK;
        float a = bk[c2];
#pragma unroll 8
        for (int j = 0; j < 64; ++j) a += xp[j] * Wk[j * CQK + c2];
        kf[((size_t)b * NTOK + r) * CQK + c2] = a;
    }
}

// ---------------------------------------------------------------------------
// Kernel 2: flash attention. 4 waves/block, each wave owns a 16-query tile.
// Scores via VALU (d=8), P laid out per-lane to match the f16 A-matrix layout,
// P@V via v_wmma_f32_16x16x32_f16 accumulating a 16x64 tile.
// ---------------------------------------------------------------------------
__global__ __launch_bounds__(128)
void attention_kernel(const float* __restrict__ qf, const float* __restrict__ kf,
                      const _Float16* __restrict__ vT, float* __restrict__ attended)
{
    __shared__ __align__(16) float kt[32 * CQK];   // shared K tile (32 keys x 8 ch)

    const int b     = blockIdx.x >> 6;             // grid = B*64 blocks
    const int qbase = (blockIdx.x & 63) * 64;
    const int wave  = threadIdx.x >> 5;
    const int lane  = threadIdx.x & 31;
    const int lo    = lane & 15;
    const int hi    = lane >> 4;
    const int q0    = qbase + wave * 16;

    // each lane carries query row (q0+lo), pre-scaled by 1/sqrt(8)
    float qreg[8];
    {
        const float* qp = qf + ((size_t)b * NTOK + q0 + lo) * CQK;
        const float scale = 0.35355339059327f;
#pragma unroll
        for (int j = 0; j < 8; ++j) qreg[j] = qp[j] * scale;
    }

    v8f acc[4];
#pragma unroll
    for (int tt = 0; tt < 4; ++tt)
#pragma unroll
        for (int g = 0; g < 8; ++g) acc[tt][g] = 0.f;

    float mrow = -__builtin_inff();
    float lrow = 0.f;

    const float*    kbase = kf + (size_t)b * NTOK * CQK;
    const _Float16* vbase = vT + ((size_t)b * CCH) * NTOK;

    for (int key0 = 0; key0 < NTOK; key0 += 32) {
        __syncthreads();
        // stage 32 keys x 8 f32 (1KB) into LDS: all 128 threads, float2 each
        ((float2*)kt)[threadIdx.x] = ((const float2*)(kbase + key0 * CQK))[threadIdx.x];
        // speculative prefetch of next K tile (global_prefetch path)
        __builtin_prefetch(kbase + (key0 + 32) * CQK + threadIdx.x * 2, 0, 0);
        __syncthreads();

        // 16 scores per lane; key index = key0 + hi*16 + g*8 + j
        float s[16];
#pragma unroll
        for (int g = 0; g < 2; ++g) {
#pragma unroll
            for (int j = 0; j < 8; ++j) {
                const float* kv = kt + (hi * 16 + g * 8 + j) * CQK;
                s[g * 8 + j] = qreg[0] * kv[0] + qreg[1] * kv[1] + qreg[2] * kv[2] + qreg[3] * kv[3]
                             + qreg[4] * kv[4] + qreg[5] * kv[5] + qreg[6] * kv[6] + qreg[7] * kv[7];
            }
        }

        // online softmax stats (row = lo, split across lane pair {lo, lo+16})
        float tmax = s[0];
#pragma unroll
        for (int kk = 1; kk < 16; ++kk) tmax = fmaxf(tmax, s[kk]);
        tmax = fmaxf(tmax, __shfl_xor(tmax, 16));
        const float mnew = fmaxf(mrow, tmax);

        v16h pa;               // half h of pa == A-matrix K-slot h for this lane
        float lsum = 0.f;
#pragma unroll
        for (int kk = 0; kk < 16; ++kk) {
            const float p = __builtin_exp2f((s[kk] - mnew) * L2E);
            lsum += p;
            pa[kk] = (_Float16)p;
        }
        lsum += __shfl_xor(lsum, 16);

        // rescale accumulators only when the running max actually moved
        // (wave-uniform branch: EXEC stays all-ones for the WMMAs below)
        if (__any(mnew > mrow)) {
            const float alpha = __builtin_exp2f((mrow - mnew) * L2E);
            lrow *= alpha;
#pragma unroll
            for (int g = 0; g < 8; ++g) {   // C/D layout row M = g + 8*hi
                const float ag = __shfl(alpha, g + 8 * hi);
                acc[0][g] *= ag; acc[1][g] *= ag; acc[2][g] *= ag; acc[3][g] *= ag;
            }
        }
        lrow += lsum;
        mrow  = mnew;

        // P[16x32] @ V[32x64] : 4 WMMA tiles of 16 output channels each.
        // B as loaded here (contiguous k order, same as pa): lane lo = column (channel), lanes<16 K=0..15, lanes>=16 K=16..31
#pragma unroll
        for (int tt = 0; tt < 4; ++tt) {
            const int ch = tt * 16 + lo;
            const v16h vb = *(const v16h*)(vbase + (size_t)ch * NTOK + key0 + hi * 16);
            acc[tt] = __builtin_amdgcn_wmma_f32_16x16x32_f16(
                false, pa, false, vb, (short)0, acc[tt], false, false);
        }
    }

    // normalize and store attended[b, q0+M, ch]
    const float inv = __builtin_amdgcn_rcpf(lrow);
    float* obase = attended + ((size_t)b * NTOK + q0) * CCH;
#pragma unroll
    for (int g = 0; g < 8; ++g) {
        const float ig = __shfl(inv, g + 8 * hi);
        const int row = g + 8 * hi;
#pragma unroll
        for (int tt = 0; tt < 4; ++tt)
            obase[(size_t)row * CCH + tt * 16 + lo] = acc[tt][g] * ig;
    }
}

// ---------------------------------------------------------------------------
// Kernel 3: out = x + gamma * upsample(attended). float4 streaming pass.
// ---------------------------------------------------------------------------
__global__ __launch_bounds__(256)
void add_upsample_kernel(const float* __restrict__ x, const float* __restrict__ gamma,
                         const float* __restrict__ attended, float* __restrict__ out)
{
    const size_t idx = (size_t)blockIdx.x * blockDim.x + threadIdx.x;  // float4 index
    const float g = gamma[0];
    const size_t vox = idx >> 4;        // 16 float4 per voxel (C=64)
    const int c4 = (int)(idx & 15);
    const int d  = (int)(vox & 63);
    const int w  = (int)((vox >> 6) & 63);
    const int h  = (int)((vox >> 12) & 63);
    const int b  = (int)(vox >> 18);
    const size_t aidx = (((size_t)b * NTOK) + ((size_t)(h >> 2) * 16 + (w >> 2)) * 16 + (d >> 2)) * CCH + c4 * 4;

    const float4 xv = ((const float4*)x)[idx];
    const float4 av = *(const float4*)(attended + aidx);
    float4 o;
    o.x = xv.x + g * av.x;
    o.y = xv.y + g * av.y;
    o.z = xv.z + g * av.z;
    o.w = xv.w + g * av.w;
    ((float4*)out)[idx] = o;
}

// ---------------------------------------------------------------------------
extern "C" void kernel_launch(void* const* d_in, const int* in_sizes, int n_in,
                              void* d_out, int out_size, void* d_ws, size_t ws_size,
                              hipStream_t stream)
{
    const float* x     = (const float*)d_in[0];
    const float* Wq    = (const float*)d_in[1];
    const float* bq    = (const float*)d_in[2];
    const float* Wk    = (const float*)d_in[3];
    const float* bk    = (const float*)d_in[4];
    const float* Wv    = (const float*)d_in[5];
    const float* bv    = (const float*)d_in[6];
    const float* gamma = (const float*)d_in[7];
    float* out = (float*)d_out;

    // workspace layout
    char* ws = (char*)d_ws;
    float*    qf       = (float*)(ws);                              // B*4096*8 f32  (256 KB)
    float*    kf       = (float*)(ws + (size_t)256 * 1024);         // B*4096*8 f32  (256 KB)
    _Float16* vT       = (_Float16*)(ws + (size_t)512 * 1024);      // B*64*4096 f16 (1 MB)
    float*    attended = (float*)(ws + (size_t)1536 * 1024);        // B*4096*64 f32 (2 MB)

    // 1) pool + project: one block per pooled token
    pool_project_kernel<<<BB * NTOK, 64, 0, stream>>>(x, Wq, bq, Wk, bk, Wv, bv, qf, kf, vT);

    // 2) flash attention: B*64 blocks x 128 threads (4 waves, 16 queries each)
    attention_kernel<<<BB * 64, 128, 0, stream>>>(qf, kf, vT, attended);

    // 3) residual + nearest upsample: B*64^3*16 float4 elements
    const size_t nvec4 = (size_t)BB * 64 * 64 * 64 * (CCH / 4);
    add_upsample_kernel<<<(unsigned)(nvec4 / 256), 256, 0, stream>>>(x, gamma, attended, out);
}
